// MultiHeadLatentAttentionNaive_64501818851862
// MI455X (gfx1250) — matrix operand order repaired and checked by reading, not compiled, not run on hardware
//
#include <hip/hip_runtime.h>
#include <hip/hip_bf16.h>
#include <math.h>

// ---------------------------------------------------------------------------
// MLA forward for MI455X (gfx1250, wave32, WMMA).
//   1) f32 -> bf16 converts (x + 4 weights)
//   2) q = x@Wq, latent = x@Wdkv, kv = latent@Wukv   (double-buffered WMMA GEMM)
//   3) ctx = causal flash-attention (WMMA, online softmax, scores never in HBM)
//   4) out = ctx@Wo + b (f32)
// Async global->LDS staging (ASYNCcnt); LDS fragment loads software-pipelined.
// ---------------------------------------------------------------------------

typedef __attribute__((ext_vector_type(16))) __bf16 bf16x16;
typedef __attribute__((ext_vector_type(8)))  float  f32x8;
typedef __attribute__((ext_vector_type(4)))  int    i32x4;

#define D_IN      2048
#define D_OUT     2048
#define NUM_HEADS 16
#define HEAD_DIM  128
#define LATENT    256
#define BATCH     2
#define SEQ       2048
#define MTOK      (BATCH * SEQ)          // 4096 flattened tokens

// ---- gfx1250 async global->LDS copy (guarded; falls back to sync copy) ----
#if defined(__has_builtin)
#if __has_builtin(__builtin_amdgcn_global_load_async_to_lds_b128) && \
    __has_builtin(__builtin_amdgcn_s_wait_asynccnt)
#define USE_ASYNC_COPY 1
#endif
#endif
#ifndef USE_ASYNC_COPY
#define USE_ASYNC_COPY 0
#endif

#if USE_ASYNC_COPY
typedef __attribute__((address_space(1))) i32x4 g_i32x4;   // global int4
typedef __attribute__((address_space(3))) i32x4 l_i32x4;   // LDS int4
#endif

__device__ __forceinline__ void copy16_to_lds(__bf16* lds_dst, const __bf16* gsrc) {
#if USE_ASYNC_COPY
  __builtin_amdgcn_global_load_async_to_lds_b128(
      (g_i32x4*)(gsrc), (l_i32x4*)(lds_dst), 0, 0);
#else
  *reinterpret_cast<uint4*>(lds_dst) = *reinterpret_cast<const uint4*>(gsrc);
#endif
}

__device__ __forceinline__ void async_wait_all() {
#if USE_ASYNC_COPY
  __builtin_amdgcn_s_wait_asynccnt(0);
#endif
}

// -------------------------------- convert ---------------------------------
__global__ __launch_bounds__(256) void cvt_f32_bf16(const float* __restrict__ src,
                                                    __bf16* __restrict__ dst, int n) {
  int i = (blockIdx.x * 256 + threadIdx.x) * 4;
  if (i + 3 >= n) return;
  float4 v = *reinterpret_cast<const float4*>(src + i);
  __bf16 o[4];
  o[0] = (__bf16)v.x; o[1] = (__bf16)v.y; o[2] = (__bf16)v.z; o[3] = (__bf16)v.w;
  *reinterpret_cast<uint2*>(dst + i) = *reinterpret_cast<uint2*>(o);
}

// ------------------------------ bf16 GEMM ---------------------------------
// C[M,N] = A[M,K] @ B[K,N].  Block tile 64x256, 8 waves, wave tile 32x64.
// Double-buffered LDS, single barrier per k-step, B-frag loads ping-ponged.
// A-frag (16-bit A 16x32): lane l -> row l%16, h=l/16; elems 0..15 =
//   contiguous K = h*16..+15.  B-frag (32x16): lane l -> col l%16,
//   contiguous K = (l/16)*16..+15.  C/D: lane->col l%16, VGPR r -> row r+8*(l/16).
template <bool OUT_BF16>
__global__ __launch_bounds__(256)
void gemm_bf16_wmma(const __bf16* __restrict__ A, const __bf16* __restrict__ B,
                    void* __restrict__ Cout, const float* __restrict__ bias,
                    int M, int N, int K) {
  constexpr int SA = 40;   // padded K-stride (bf16), keeps 16B alignment
  constexpr int SB = 40;
  __shared__ __bf16 sA[2][64 * SA];      // 2 x  5120 B
  __shared__ __bf16 sBt[2][256 * SB];    // 2 x 20480 B

  const int tid  = threadIdx.x;
  const int lane = tid & 31;
  const int wave = tid >> 5;
  const int wr   = wave >> 2;           // 0..1 (32-row tile)
  const int wc   = wave & 3;            // 0..3 (64-col tile)
  const int lm   = lane & 15;
  const int lh   = lane >> 4;
  const int m0   = blockIdx.y * 64;
  const int n0   = blockIdx.x * 256;

  const int ar = tid >> 2;              // A stage: row 0..63
  const int ak = (tid & 3) * 8;         //          k-part {0,8,16,24}
  const int kp = tid >> 4;              // B stage: k-pair 0..15 -> k = 2*kp
  const int nb = (tid & 15) * 16;       //          n-part {0..240}

  // stage tile ks (k-offset ks*32) into buffer buf
  auto stage = [&](int ks, int buf) {
    // A tile: straight row-major copy -> async to LDS
    copy16_to_lds(&sA[buf][ar * SA + ak],
                  &A[(size_t)(m0 + ar) * K + ks * 32 + ak]);
    // B tile transposed: two k-rows x 16 cols, packed b32 pair stores
    const __bf16* g0 = &B[(size_t)(ks * 32 + 2 * kp) * N + n0 + nb];
    const __bf16* g1 = g0 + N;
    __bf16 t0[16], t1[16];
    *reinterpret_cast<uint4*>(t0)     = *reinterpret_cast<const uint4*>(g0);
    *reinterpret_cast<uint4*>(t0 + 8) = *reinterpret_cast<const uint4*>(g0 + 8);
    *reinterpret_cast<uint4*>(t1)     = *reinterpret_cast<const uint4*>(g1);
    *reinterpret_cast<uint4*>(t1 + 8) = *reinterpret_cast<const uint4*>(g1 + 8);
#pragma unroll
    for (int i = 0; i < 16; ++i) {
      __bf16 pr[2] = {t0[i], t1[i]};
      *reinterpret_cast<unsigned int*>(&sBt[buf][(nb + i) * SB + 2 * kp]) =
          *reinterpret_cast<unsigned int*>(pr);
    }
  };

  f32x8 acc[2][4] = {};
  const int nk = K / 32;

  stage(0, 0);
  for (int ki = 0; ki < nk; ++ki) {
    const int cur = ki & 1;
    async_wait_all();
    __syncthreads();
    if (ki + 1 < nk) stage(ki + 1, cur ^ 1);

    bf16x16 af[2];
#pragma unroll
    for (int mt = 0; mt < 2; ++mt) {
      const int row = wr * 32 + mt * 16 + lm;
      uint4* p = reinterpret_cast<uint4*>(&af[mt]);
      p[0] = *reinterpret_cast<const uint4*>(&sA[cur][row * SA + lh * 16]);
      p[1] = *reinterpret_cast<const uint4*>(&sA[cur][row * SA + lh * 16 + 8]);
    }
    // B fragments ping-ponged: load nt+1 before WMMAs of nt
    bf16x16 bb[2];
    auto load_b = [&](int nt, int slot) {
      const int col = wc * 64 + nt * 16 + lm;
      uint4* p = reinterpret_cast<uint4*>(&bb[slot]);
      p[0] = *reinterpret_cast<const uint4*>(&sBt[cur][col * SB + lh * 16]);
      p[1] = *reinterpret_cast<const uint4*>(&sBt[cur][col * SB + lh * 16 + 8]);
    };
    load_b(0, 0);
#pragma unroll
    for (int nt = 0; nt < 4; ++nt) {
      if (nt + 1 < 4) load_b(nt + 1, (nt + 1) & 1);
#pragma unroll
      for (int mt = 0; mt < 2; ++mt)
        acc[mt][nt] = __builtin_amdgcn_wmma_f32_16x16x32_bf16(
            false, af[mt], false, bb[nt & 1], (short)0, acc[mt][nt], false, false);
    }
  }

  // ---- epilogue ----
#pragma unroll
  for (int mt = 0; mt < 2; ++mt) {
#pragma unroll
    for (int nt = 0; nt < 4; ++nt) {
      const int gc = n0 + wc * 64 + nt * 16 + lm;
#pragma unroll
      for (int r = 0; r < 8; ++r) {
        const int gr = m0 + wr * 32 + mt * 16 + r + 8 * lh;
        const float v = acc[mt][nt][r];
        if (OUT_BF16) {
          reinterpret_cast<__bf16*>(Cout)[(size_t)gr * N + gc] = (__bf16)v;
        } else {
          reinterpret_cast<float*>(Cout)[(size_t)gr * N + gc] = v + bias[gc];
        }
      }
    }
  }
}

// --------------------------- flash attention ------------------------------
// One block = (batch b, head h, 128 query rows). 8 waves x 16 q-rows each.
// Key blocks of 64 streamed through LDS; online softmax; O accum 16x128 f32/wave.
__global__ __launch_bounds__(256)
void mla_flash_attn(const __bf16* __restrict__ qb, const __bf16* __restrict__ kvb,
                    __bf16* __restrict__ ctx) {
  constexpr int SK = 136;  // sK  [64 tok][128 d], padded d-stride
  constexpr int SV = 72;   // sVt [128 d][64 tok], padded t-stride
  constexpr int SP = 72;   // sP  per-wave [16 rows][64 tok], padded
  __shared__ __bf16 sK[64 * SK];
  __shared__ __bf16 sVt[128 * SV];
  __shared__ __bf16 sP[8 * 16 * SP];

  const int tid  = threadIdx.x;
  const int lane = tid & 31;
  const int wave = tid >> 5;
  const int lm   = lane & 15;
  const int lh   = lane >> 4;
  const int q0   = blockIdx.x * 128;
  const int head = blockIdx.y;
  const int b    = blockIdx.z;

  const size_t qbase  = (size_t)b * SEQ * D_OUT + (size_t)head * HEAD_DIM;
  const size_t kvbase = (size_t)b * SEQ * (2 * D_OUT) + (size_t)head * HEAD_DIM;
  const size_t vbase  = kvbase + D_OUT;

  // resident q fragments (A layout), rows q0 + wave*16 + (lane%16)
  bf16x16 qf[4];
  {
    const __bf16* qrow = qb + qbase + (size_t)(q0 + wave * 16 + lm) * D_OUT;
#pragma unroll
    for (int kc = 0; kc < 4; ++kc) {
      uint4* p = reinterpret_cast<uint4*>(&qf[kc]);
      p[0] = *reinterpret_cast<const uint4*>(qrow + kc * 32 + lh * 16);
      p[1] = *reinterpret_cast<const uint4*>(qrow + kc * 32 + lh * 16 + 8);
    }
  }

  f32x8 O[8] = {};
  float mrun[8], lrun[8];
#pragma unroll
  for (int r = 0; r < 8; ++r) { mrun[r] = -1e30f; lrun[r] = 0.f; }

  const float isd = 0.08838834764831845f;   // rsqrt(128)

  const int st = tid >> 2;            // K staging token row 0..63
  const int sd = (tid & 3) * 32;      // K staging d-part
  const int tp = tid >> 3;            // V staging token pair 0..31
  const int dp = (tid & 7) * 16;      // V staging d-part {0..112}

  const int kb_end = q0 / 64 + 2;     // causal: last key <= q0+127
  for (int kb = 0; kb < kb_end; ++kb) {
    const int kt0 = kb * 64;
    // ---- stage K row-major (async global->LDS) ----
    {
      const __bf16* src = kvb + kvbase + (size_t)(kt0 + st) * (2 * D_OUT) + sd;
#pragma unroll
      for (int j = 0; j < 4; ++j)
        copy16_to_lds(&sK[st * SK + sd + j * 8], src + j * 8);
    }
    // ---- stage V transposed: two tokens/thread, packed b32 stores ----
    {
      const __bf16* v0 = kvb + vbase + (size_t)(kt0 + 2 * tp) * (2 * D_OUT) + dp;
      const __bf16* v1 = v0 + 2 * D_OUT;
      __bf16 t0[16], t1[16];
      *reinterpret_cast<uint4*>(t0)     = *reinterpret_cast<const uint4*>(v0);
      *reinterpret_cast<uint4*>(t0 + 8) = *reinterpret_cast<const uint4*>(v0 + 8);
      *reinterpret_cast<uint4*>(t1)     = *reinterpret_cast<const uint4*>(v1);
      *reinterpret_cast<uint4*>(t1 + 8) = *reinterpret_cast<const uint4*>(v1 + 8);
#pragma unroll
      for (int i = 0; i < 16; ++i) {
        __bf16 pr[2] = {t0[i], t1[i]};
        *reinterpret_cast<unsigned int*>(&sVt[(dp + i) * SV + 2 * tp]) =
            *reinterpret_cast<unsigned int*>(pr);
      }
    }
    async_wait_all();
    __syncthreads();

    // ---- scores: linear idx (kc major), K-frag loads ping-ponged ----
    f32x8 sc[4] = {};
    {
      bf16x16 kf[2];
      auto load_kf = [&](int idx, int slot) {
        const int kc = idx >> 2, nt = idx & 3;
        const int krow = nt * 16 + lm;
        uint4* p = reinterpret_cast<uint4*>(&kf[slot]);
        p[0] = *reinterpret_cast<const uint4*>(&sK[krow * SK + kc * 32 + lh * 16]);
        p[1] = *reinterpret_cast<const uint4*>(&sK[krow * SK + kc * 32 + lh * 16 + 8]);
      };
      load_kf(0, 0);
#pragma unroll
      for (int idx = 0; idx < 16; ++idx) {
        if (idx + 1 < 16) load_kf(idx + 1, (idx + 1) & 1);
        const int kc = idx >> 2, nt = idx & 3;
        sc[nt] = __builtin_amdgcn_wmma_f32_16x16x32_bf16(
            false, qf[kc], false, kf[idx & 1], (short)0, sc[nt], false, false);
      }
    }

    // ---- causal mask (before scale, per reference) + logits ----
    float lg[4][8];
#pragma unroll
    for (int nt = 0; nt < 4; ++nt) {
      const int kpos = kt0 + nt * 16 + lm;
#pragma unroll
      for (int r = 0; r < 8; ++r) {
        const int qpos = q0 + wave * 16 + r + 8 * lh;
        lg[nt][r] = (kpos <= qpos) ? sc[nt][r] * isd : -1e30f;
      }
    }

    // ---- online softmax update ----
#pragma unroll
    for (int r = 0; r < 8; ++r) {
      float mx = fmaxf(fmaxf(lg[0][r], lg[1][r]), fmaxf(lg[2][r], lg[3][r]));
#pragma unroll
      for (int off = 8; off >= 1; off >>= 1) mx = fmaxf(mx, __shfl_xor(mx, off, 32));
      const float mnew  = fmaxf(mrun[r], mx);
      const float scale = __expf(mrun[r] - mnew);
      float ls = 0.f;
#pragma unroll
      for (int nt = 0; nt < 4; ++nt) {
        const float p = __expf(lg[nt][r] - mnew);
        lg[nt][r] = p;
        ls += p;
      }
#pragma unroll
      for (int off = 8; off >= 1; off >>= 1) ls += __shfl_xor(ls, off, 32);
      lrun[r] = lrun[r] * scale + ls;
      mrun[r] = mnew;
#pragma unroll
      for (int on = 0; on < 8; ++on) O[on][r] *= scale;
    }

    // ---- P (C layout) -> LDS -> A-layout fragments (wave-private region) ----
#pragma unroll
    for (int nt = 0; nt < 4; ++nt)
#pragma unroll
      for (int r = 0; r < 8; ++r)
        sP[(wave * 16 + r + 8 * lh) * SP + nt * 16 + lm] = (__bf16)lg[nt][r];

    bf16x16 pf[2];
#pragma unroll
    for (int pkc = 0; pkc < 2; ++pkc) {
      const int prow = wave * 16 + lm;
      uint4* p = reinterpret_cast<uint4*>(&pf[pkc]);
      p[0] = *reinterpret_cast<const uint4*>(&sP[prow * SP + pkc * 32 + lh * 16]);
      p[1] = *reinterpret_cast<const uint4*>(&sP[prow * SP + pkc * 32 + lh * 16 + 8]);
    }

    // ---- O += P @ V : linear idx (pkc major), V-frag loads ping-ponged ----
    {
      bf16x16 vf[2];
      auto load_vf = [&](int idx, int slot) {
        const int pkc = idx >> 3, on = idx & 7;
        const int vrow = on * 16 + lm;
        uint4* p = reinterpret_cast<uint4*>(&vf[slot]);
        p[0] = *reinterpret_cast<const uint4*>(&sVt[vrow * SV + pkc * 32 + lh * 16]);
        p[1] = *reinterpret_cast<const uint4*>(&sVt[vrow * SV + pkc * 32 + lh * 16 + 8]);
      };
      load_vf(0, 0);
#pragma unroll
      for (int idx = 0; idx < 16; ++idx) {
        if (idx + 1 < 16) load_vf(idx + 1, (idx + 1) & 1);
        const int pkc = idx >> 3, on = idx & 7;
        O[on] = __builtin_amdgcn_wmma_f32_16x16x32_bf16(
            false, pf[pkc], false, vf[idx & 1], (short)0, O[on], false, false);
      }
    }
    __syncthreads();
  }

  // ---- normalize + store ctx (heads re-merged: [b, s, h*128+d]) ----
#pragma unroll
  for (int r = 0; r < 8; ++r) {
    const float inv = 1.0f / lrun[r];
    const int qrow = q0 + wave * 16 + r + 8 * lh;
    __bf16* dst = ctx + qbase + (size_t)qrow * D_OUT;
#pragma unroll
    for (int on = 0; on < 8; ++on)
      dst[on * 16 + lm] = (__bf16)(O[on][r] * inv);
  }
}

// -------------------------------- launch ----------------------------------
extern "C" void kernel_launch(void* const* d_in, const int* in_sizes, int n_in,
                              void* d_out, int out_size, void* d_ws, size_t ws_size,
                              hipStream_t stream) {
  const float* x     = (const float*)d_in[0];
  const float* Wq    = (const float*)d_in[1];
  const float* Wdkv  = (const float*)d_in[2];
  const float* Wukv  = (const float*)d_in[3];
  const float* Wo    = (const float*)d_in[4];
  const float* bias  = (const float*)d_in[5];
  float* out = (float*)d_out;
  (void)in_sizes; (void)n_in; (void)out_size; (void)ws_size;

  __bf16* ws = (__bf16*)d_ws;
  size_t off = 0;
  auto carve = [&](size_t n) { __bf16* p = ws + off; off += n; return p; };
  __bf16* xb    = carve((size_t)MTOK * D_IN);          // 8M
  __bf16* Wqb   = carve((size_t)D_IN * D_OUT);         // 4M
  __bf16* Wdkvb = carve((size_t)D_IN * LATENT);        // 0.5M
  __bf16* Wukvb = carve((size_t)LATENT * 2 * D_OUT);   // 1M
  __bf16* Wob   = carve((size_t)D_OUT * D_IN);         // 4M
  __bf16* qbuf  = carve((size_t)MTOK * D_OUT);         // 8M
  __bf16* latb  = carve((size_t)MTOK * LATENT);        // 1M
  __bf16* kvbuf = carve((size_t)MTOK * 2 * D_OUT);     // 16M
  __bf16* ctxb  = carve((size_t)MTOK * D_OUT);         // 8M

  auto cvt = [&](const float* s, __bf16* d, int n) {
    cvt_f32_bf16<<<n / 1024, 256, 0, stream>>>(s, d, n);
  };
  cvt(x,    xb,    MTOK * D_IN);
  cvt(Wq,   Wqb,   D_IN * D_OUT);
  cvt(Wdkv, Wdkvb, D_IN * LATENT);
  cvt(Wukv, Wukvb, LATENT * 2 * D_OUT);
  cvt(Wo,   Wob,   D_OUT * D_IN);

  // q = x @ Wq                       [4096 x 2048 x 2048]
  gemm_bf16_wmma<true><<<dim3(D_OUT / 256, MTOK / 64), 256, 0, stream>>>(
      xb, Wqb, qbuf, nullptr, MTOK, D_OUT, D_IN);
  // latent = x @ Wdkv                [4096 x 256 x 2048]
  gemm_bf16_wmma<true><<<dim3(LATENT / 256, MTOK / 64), 256, 0, stream>>>(
      xb, Wdkvb, latb, nullptr, MTOK, LATENT, D_IN);
  // kv = latent @ Wukv               [4096 x 4096 x 256]
  gemm_bf16_wmma<true><<<dim3((2 * D_OUT) / 256, MTOK / 64), 256, 0, stream>>>(
      latb, Wukvb, kvbuf, nullptr, MTOK, 2 * D_OUT, LATENT);
  // ctx = flash attention
  mla_flash_attn<<<dim3(SEQ / 128, NUM_HEADS, BATCH), 256, 0, stream>>>(
      qbuf, kvbuf, ctxb);
  // out = ctx @ Wo + b               [4096 x 2048 x 2048], f32 out
  gemm_bf16_wmma<false><<<dim3(D_IN / 256, MTOK / 64), 256, 0, stream>>>(
      ctxb, Wob, out, bias, MTOK, D_IN, D_OUT);
}
